// GLFMSABlock_69131793596507
// MI455X (gfx1250) — matrix operand order repaired and checked by reading, not compiled, not run on hardware
//
#include <hip/hip_runtime.h>
#include <hip/hip_bf16.h>

typedef __bf16 v16bf __attribute__((ext_vector_type(16)));
typedef __bf16 v8bf  __attribute__((ext_vector_type(8)));
typedef float  v8f   __attribute__((ext_vector_type(8)));

#define Bc   4
#define Sc   2048
#define Dc   512
#define D3c  1536
#define Hc   2
#define HDc  256
#define BSc  (Bc * Sc)

// ---------------- WMMA fragment loaders (CDNA5 16x16x32 bf16 layouts) --------

// A-matrix 16x32 bf16: lane l (0..15) holds row M=l, K = {0..7, 16..23};
// lane l+16 holds row M=l, K = {8..15, 24..31}.  Two 16B contiguous loads.
static __device__ inline v16bf load_fragA(const __bf16* base, int ld) {
    const int lane = threadIdx.x & 31;
    const int r15  = lane & 15;
    const int hi   = lane >> 4;
    const __bf16* p = base + (size_t)r15 * ld;
    union { v16bf v; v8bf h[2]; } u;
    u.h[0] = *(const v8bf*)(p + hi * 8);
    u.h[1] = *(const v8bf*)(p + 16 + hi * 8);
    return u.v;
}

// B-matrix 32x16 bf16: lane l holds column N=(l&15); lanes 0-15 K={0..7,16..23},
// lanes 16-31 K={8..15,24..31}.  Source rows are B^T rows (row-major).
static __device__ inline v16bf load_fragB(const __bf16* base, int ld) {
    const int lane = threadIdx.x & 31;
    const int r15  = lane & 15;
    const int hi   = lane >> 4;
    const __bf16* p = base + (size_t)r15 * ld;
    union { v16bf v; v8bf h[2]; } u;
    u.h[0] = *(const v8bf*)(p + hi * 8);
    u.h[1] = *(const v8bf*)(p + 16 + hi * 8);
    return u.v;
}

#define WMMA_BF16(a, b, c) \
    __builtin_amdgcn_wmma_f32_16x16x32_bf16(false, (a), false, (b), (short)0, (c), false, false)

// ---------------- fp32 -> bf16 conversion -----------------------------------

__global__ void f32_to_bf16(const float* __restrict__ s, __bf16* __restrict__ d, size_t n) {
    size_t i  = (size_t)blockIdx.x * blockDim.x + threadIdx.x;
    size_t st = (size_t)gridDim.x * blockDim.x;
    for (; i < n; i += st) d[i] = (__bf16)s[i];
}

// ---------------- generic GEMM: C[M,N] = A[M,K] * Bw[N,K]^T + bias ----------
// A row-major bf16 (lda=K), Bw row-major bf16 (ld=K).  One wave per 64x64
// C-tile (4x4 register blocking: 16 wmma per 32-K step, 32 FLOP/B of loads).
// 8 M-adjacent waves per 256-thread block share B cache lines.

__global__ __launch_bounds__(256) void gemm_bf16_wmma(
    const __bf16* __restrict__ A, const __bf16* __restrict__ Bw,
    const float* __restrict__ bias, __bf16* __restrict__ C,
    int M, int N, int K, int ldc, int coff, int relu)
{
    const int wave = threadIdx.x >> 5;
    const int lane = threadIdx.x & 31;
    const int r15  = lane & 15;
    const int hi   = lane >> 4;
    const int tiles64M = M >> 6;
    const int t  = blockIdx.x * 8 + wave;
    const int tm = t % tiles64M;
    const int tn = t / tiles64M;

    const __bf16* Ab = A  + (size_t)(tm * 64) * K;
    const __bf16* Bb = Bw + (size_t)(tn * 64) * K;

    v8f acc[4][4];
#pragma unroll
    for (int i = 0; i < 4; ++i)
#pragma unroll
        for (int j = 0; j < 4; ++j) {
            v8f z = {0.f, 0.f, 0.f, 0.f, 0.f, 0.f, 0.f, 0.f};
            acc[i][j] = z;
        }

    for (int k0 = 0; k0 < K; k0 += 32) {
        v16bf af[4], bf[4];
#pragma unroll
        for (int i = 0; i < 4; ++i) af[i] = load_fragA(Ab + (size_t)(i * 16) * K + k0, K);
#pragma unroll
        for (int j = 0; j < 4; ++j) bf[j] = load_fragB(Bb + (size_t)(j * 16) * K + k0, K);
#pragma unroll
        for (int i = 0; i < 4; ++i)
#pragma unroll
            for (int j = 0; j < 4; ++j)
                acc[i][j] = WMMA_BF16(af[i], bf[j], acc[i][j]);
    }

#pragma unroll
    for (int j = 0; j < 4; ++j) {
        const float bcol = bias[tn * 64 + j * 16 + r15];
#pragma unroll
        for (int i = 0; i < 4; ++i) {
            __bf16* Cb = C + (size_t)(tm * 64 + i * 16 + 8 * hi) * ldc
                           + coff + tn * 64 + j * 16 + r15;
#pragma unroll
            for (int rr = 0; rr < 8; ++rr) {
                float v = acc[i][j][rr] + bcol;
                if (relu) v = fmaxf(v, 0.f);
                Cb[(size_t)rr * ldc] = (__bf16)v;
            }
        }
    }
}

// ---------------- global causal attention: flash-style, WMMA ----------------
// One wave per (b, h, 16-query block).  hd = 256.  Streams key blocks of 32
// with online softmax; Q*K^T and P*V on v_wmma_f32_16x16x32_bf16.

__global__ __launch_bounds__(32) void flash_causal(
    const __bf16* __restrict__ qkv, __bf16* __restrict__ out)
{
    __shared__ __bf16 vt[HDc * 32];   // V^T tile: [hd 256][keys 32]
    __shared__ __bf16 pt[16 * 32];    // P tile:   [rows 16][keys 32]

    const int qb = blockIdx.x;
    const int h  = blockIdx.y;
    const int b  = blockIdx.z;
    const int lane = threadIdx.x & 31;
    const int r15  = lane & 15;
    const int hi   = lane >> 4;

    const __bf16* qbase = qkv + ((size_t)(b * Sc + qb * 16)) * D3c + h * HDc;
    v16bf qf[8];
#pragma unroll
    for (int i = 0; i < 8; ++i) qf[i] = load_fragA(qbase + i * 32, D3c);

    float m[8], lsum[8], sfv[8];
    v8f o[16];
#pragma unroll
    for (int rr = 0; rr < 8; ++rr) { m[rr] = -1e30f; lsum[rr] = 0.f; }
#pragma unroll
    for (int j = 0; j < 16; ++j) { v8f z = {0.f,0.f,0.f,0.f,0.f,0.f,0.f,0.f}; o[j] = z; }

    const float scale = 0.0625f;  // 1/sqrt(256)
    const int nkb = qb / 2 + 1;   // 32-key blocks; diagonal handled by mask

    for (int kb = 0; kb < nkb; ++kb) {
        const int key0 = kb * 32;

        // ---- stage V^T in LDS: lane owns key row `lane`
        const __bf16* vbase = qkv + ((size_t)(b * Sc + key0 + lane)) * D3c + 2 * Dc + h * HDc;
        for (int n0 = 0; n0 < HDc; n0 += 8) {
            v8bf vv = *(const v8bf*)(vbase + n0);
#pragma unroll
            for (int e = 0; e < 8; ++e) vt[(n0 + e) * 32 + lane] = vv[e];
        }
        __syncthreads();

        // ---- scores: two 16x16 tiles over hd=256 (8 wmma each)
        v8f s0 = {0.f,0.f,0.f,0.f,0.f,0.f,0.f,0.f};
        v8f s1 = {0.f,0.f,0.f,0.f,0.f,0.f,0.f,0.f};
        const __bf16* kbase0 = qkv + ((size_t)(b * Sc + key0)) * D3c + Dc + h * HDc;
        const __bf16* kbase1 = kbase0 + (size_t)16 * D3c;
#pragma unroll
        for (int i = 0; i < 8; ++i) s0 = WMMA_BF16(qf[i], load_fragB(kbase0 + i * 32, D3c), s0);
#pragma unroll
        for (int i = 0; i < 8; ++i) s1 = WMMA_BF16(qf[i], load_fragB(kbase1 + i * 32, D3c), s1);

        // ---- causal mask + scale + online softmax (rows spread over D layout)
        const int colg0 = key0 + r15;
        const int colg1 = key0 + 16 + r15;
#pragma unroll
        for (int rr = 0; rr < 8; ++rr) {
            const int rowg = qb * 16 + rr + 8 * hi;
            s0[rr] = (colg0 > rowg) ? -1e30f : s0[rr] * scale;
            s1[rr] = (colg1 > rowg) ? -1e30f : s1[rr] * scale;

            float mx = fmaxf(s0[rr], s1[rr]);
#pragma unroll
            for (int d = 1; d < 16; d <<= 1) mx = fmaxf(mx, __shfl_xor(mx, d, 32));
            const float mnew = fmaxf(m[rr], mx);
            const float sf   = __expf(m[rr] - mnew);
            const float p0   = __expf(s0[rr] - mnew);
            const float p1   = __expf(s1[rr] - mnew);
            float rs = p0 + p1;
#pragma unroll
            for (int d = 1; d < 16; d <<= 1) rs += __shfl_xor(rs, d, 32);
            lsum[rr] = lsum[rr] * sf + rs;
            m[rr]    = mnew;
            sfv[rr]  = sf;
            s0[rr] = p0;
            s1[rr] = p1;
        }

        // ---- rescale accumulators
#pragma unroll
        for (int j = 0; j < 16; ++j)
#pragma unroll
            for (int rr = 0; rr < 8; ++rr) o[j][rr] *= sfv[rr];

        // ---- P: D-layout -> A-layout via LDS
#pragma unroll
        for (int rr = 0; rr < 8; ++rr) {
            pt[(rr + 8 * hi) * 32 + r15]      = (__bf16)s0[rr];
            pt[(rr + 8 * hi) * 32 + 16 + r15] = (__bf16)s1[rr];
        }
        __syncthreads();
        v16bf pa = load_fragA(pt, 32);

        // ---- O += P * V  (16 wmma across hd)
#pragma unroll
        for (int j = 0; j < 16; ++j)
            o[j] = WMMA_BF16(pa, load_fragB(&vt[(j * 16) * 32], 32), o[j]);
        __syncthreads();
    }

    // ---- normalize and write out (row-major [B*S, D])
    __bf16* obase = out + ((size_t)(b * Sc + qb * 16)) * Dc + h * HDc;
#pragma unroll
    for (int j = 0; j < 16; ++j)
#pragma unroll
        for (int rr = 0; rr < 8; ++rr) {
            const float v = o[j][rr] / lsum[rr];
            obase[(size_t)(rr + 8 * hi) * Dc + j * 16 + r15] = (__bf16)v;
        }
}

// ---------------- local banded attention (WIN=2 -> 5 keys/query) ------------

__global__ __launch_bounds__(256) void local_attn(
    const __bf16* __restrict__ qkv, __bf16* __restrict__ out)
{
    const int idx = blockIdx.x * 256 + threadIdx.x;
    if (idx >= Bc * Hc * Sc) return;
    const int q = idx & (Sc - 1);
    const int h = (idx >> 11) & (Hc - 1);
    const int b = idx >> 12;

    const __bf16* qrow = qkv + ((size_t)(b * Sc + q)) * D3c + h * HDc;
    float sc[5];
    float mx = -1e30f;
#pragma unroll
    for (int w = 0; w < 5; ++w) {
        const int k = q + w - 2;
        if (k >= 0 && k < Sc) {
            const __bf16* krow = qkv + ((size_t)(b * Sc + k)) * D3c + Dc + h * HDc;
            float dot = 0.f;
            for (int d = 0; d < HDc; ++d) dot += (float)qrow[d] * (float)krow[d];
            sc[w] = dot * 0.0625f;
            mx = fmaxf(mx, sc[w]);
        } else sc[w] = -1e30f;
    }
    float p[5], sum = 0.f;
#pragma unroll
    for (int w = 0; w < 5; ++w) { p[w] = __expf(sc[w] - mx); sum += p[w]; }
    const float inv = 1.f / sum;

    __bf16* orow = out + ((size_t)(b * Sc + q)) * Dc + h * HDc;
    for (int d = 0; d < HDc; ++d) {
        float a = 0.f;
#pragma unroll
        for (int w = 0; w < 5; ++w) {
            const int k = q + w - 2;
            if (k >= 0 && k < Sc)
                a += p[w] * (float)qkv[((size_t)(b * Sc + k)) * D3c + 2 * Dc + h * HDc + d];
        }
        orow[d] = (__bf16)(a * inv);
    }
}

// ---------------- residual add + layernorm (writes f32 + bf16) --------------

__global__ __launch_bounds__(256) void add_ln(
    const float* __restrict__ res, const __bf16* __restrict__ y,
    const float* __restrict__ g, const float* __restrict__ be,
    float* __restrict__ outf, __bf16* __restrict__ outb)
{
    __shared__ float red[256];
    const int t = threadIdx.x;
    const size_t row = (size_t)blockIdx.x * Dc;

    float v0 = res[row + t]       + (float)y[row + t];
    float v1 = res[row + t + 256] + (float)y[row + t + 256];
    red[t] = v0 + v1;
    __syncthreads();
    for (int s2 = 128; s2 > 0; s2 >>= 1) { if (t < s2) red[t] += red[t + s2]; __syncthreads(); }
    const float mu = red[0] * (1.f / (float)Dc);
    __syncthreads();

    const float d0 = v0 - mu, d1 = v1 - mu;
    red[t] = d0 * d0 + d1 * d1;
    __syncthreads();
    for (int s2 = 128; s2 > 0; s2 >>= 1) { if (t < s2) red[t] += red[t + s2]; __syncthreads(); }
    const float rstd = rsqrtf(red[0] * (1.f / (float)Dc) + 1e-5f);

    const float o0 = d0 * rstd * g[t]       + be[t];
    const float o1 = d1 * rstd * g[t + 256] + be[t + 256];
    outf[row + t]       = o0;
    outf[row + t + 256] = o1;
    outb[row + t]       = (__bf16)o0;
    outb[row + t + 256] = (__bf16)o1;
}

// ---------------- driver ----------------------------------------------------

extern "C" void kernel_launch(void* const* d_in, const int* in_sizes, int n_in,
                              void* d_out, int out_size, void* d_ws, size_t ws_size,
                              hipStream_t stream) {
    const float* x      = (const float*)d_in[0];
    const float* gw_in  = (const float*)d_in[1];
    const float* gb_in  = (const float*)d_in[2];
    const float* gw_out = (const float*)d_in[3];
    const float* gb_out = (const float*)d_in[4];
    const float* lw_in  = (const float*)d_in[5];
    const float* lb_in  = (const float*)d_in[6];
    const float* lw_out = (const float*)d_in[7];
    const float* lb_out = (const float*)d_in[8];
    const float* fuse_w = (const float*)d_in[9];
    const float* fuse_b = (const float*)d_in[10];
    const float* ln1_g  = (const float*)d_in[11];
    const float* ln1_b  = (const float*)d_in[12];
    const float* ff_w1  = (const float*)d_in[13];
    const float* ff_b1  = (const float*)d_in[14];
    const float* ff_w2  = (const float*)d_in[15];
    const float* ff_b2  = (const float*)d_in[16];
    const float* ln2_g  = (const float*)d_in[17];
    const float* ln2_b  = (const float*)d_in[18];

    size_t off = 0;
    auto abf = [&](size_t n) { __bf16* p = (__bf16*)((char*)d_ws + off);
                               off = (off + n * 2 + 255) & ~(size_t)255; return p; };
    auto af32 = [&](size_t n) { float* p = (float*)((char*)d_ws + off);
                                off = (off + n * 4 + 255) & ~(size_t)255; return p; };

    __bf16* xb      = abf((size_t)BSc * Dc);
    __bf16* gw_inb  = abf((size_t)3 * Dc * Dc);
    __bf16* lw_inb  = abf((size_t)3 * Dc * Dc);
    __bf16* gw_outb = abf((size_t)Dc * Dc);
    __bf16* lw_outb = abf((size_t)Dc * Dc);
    __bf16* fuse_wb = abf((size_t)Dc * 2 * Dc);
    __bf16* ff_w1b  = abf((size_t)4 * Dc * Dc);
    __bf16* ff_w2b  = abf((size_t)Dc * 4 * Dc);
    __bf16* gqkv    = abf((size_t)BSc * D3c);
    __bf16* lqkv    = abf((size_t)BSc * D3c);
    __bf16* gattn   = abf((size_t)BSc * Dc);
    __bf16* lattn   = abf((size_t)BSc * Dc);
    __bf16* concatb = abf((size_t)BSc * 2 * Dc);
    __bf16* fusedb  = abf((size_t)BSc * Dc);
    __bf16* x1b     = abf((size_t)BSc * Dc);
    __bf16* hb      = abf((size_t)BSc * 4 * Dc);
    __bf16* ffb     = abf((size_t)BSc * Dc);
    float*  x1f     = af32((size_t)BSc * Dc);

    auto cvt = [&](const float* s, __bf16* d, size_t n) {
        f32_to_bf16<<<1024, 256, 0, stream>>>(s, d, n);
    };
    cvt(x,      xb,      (size_t)BSc * Dc);
    cvt(gw_in,  gw_inb,  (size_t)3 * Dc * Dc);
    cvt(lw_in,  lw_inb,  (size_t)3 * Dc * Dc);
    cvt(gw_out, gw_outb, (size_t)Dc * Dc);
    cvt(lw_out, lw_outb, (size_t)Dc * Dc);
    cvt(fuse_w, fuse_wb, (size_t)Dc * 2 * Dc);
    cvt(ff_w1,  ff_w1b,  (size_t)4 * Dc * Dc);
    cvt(ff_w2,  ff_w2b,  (size_t)Dc * 4 * Dc);

    auto gemm = [&](const __bf16* A, const __bf16* Bw, const float* bias, __bf16* C,
                    int M, int N, int K, int ldc, int coff, int relu) {
        const int blocks = (M / 64) * (N / 64) / 8;
        gemm_bf16_wmma<<<blocks, 256, 0, stream>>>(A, Bw, bias, C, M, N, K, ldc, coff, relu);
    };

    // QKV projections
    gemm(xb, gw_inb, gb_in, gqkv, BSc, 3 * Dc, Dc, D3c, 0, 0);
    gemm(xb, lw_inb, lb_in, lqkv, BSc, 3 * Dc, Dc, D3c, 0, 0);

    // attention
    flash_causal<<<dim3(Sc / 16, Hc, Bc), 32, 0, stream>>>(gqkv, gattn);
    local_attn<<<(Bc * Hc * Sc + 255) / 256, 256, 0, stream>>>(lqkv, lattn);

    // output projections into concat buffer, then fuse
    gemm(gattn, gw_outb, gb_out, concatb, BSc, Dc, Dc, 2 * Dc, 0, 0);
    gemm(lattn, lw_outb, lb_out, concatb, BSc, Dc, Dc, 2 * Dc, Dc, 0);
    gemm(concatb, fuse_wb, fuse_b, fusedb, BSc, Dc, 2 * Dc, Dc, 0, 0);

    // x1 = LN(x + fused)
    add_ln<<<BSc, 256, 0, stream>>>(x, fusedb, ln1_g, ln1_b, x1f, x1b);

    // FFN
    gemm(x1b, ff_w1b, ff_b1, hb,  BSc, 4 * Dc, Dc, 4 * Dc, 0, 1);
    gemm(hb,  ff_w2b, ff_b2, ffb, BSc, Dc, 4 * Dc, Dc, 0, 0);

    // out = LN(x1 + ff)
    add_ln<<<BSc, 256, 0, stream>>>(x1f, ffb, ln2_g, ln2_b, (float*)d_out, fusedb);
}
